// WOQ_Linear_Model_82841329206004
// MI455X (gfx1250) — matrix operand order repaired and checked by reading, not compiled, not run on hardware
//
#include <hip/hip_runtime.h>
#include <hip/hip_bf16.h>
#include <cstdint>

typedef __attribute__((ext_vector_type(16))) __bf16 v16bf;
typedef __attribute__((ext_vector_type(8)))  __bf16 v8bf;
typedef __attribute__((ext_vector_type(8)))  float  v8f;

#define OUT_F 11008
#define IN_F  4096
#define GROUP 128
#define NGRP  (IN_F / GROUP)     // 32
#define MTOK  4096

#define TILE_M 256
#define TILE_N 128
#define KSTAGE 64                // half a quant group per pipeline stage
#define NSTAGE (IN_F / KSTAGE)   // 64
#define NTHREADS 512             // 16 waves: 4 along M x 4 along N

// Each wave computes a 64(M) x 32(N) C-tile: 4 M-subtiles x 2 N-subtiles.

__global__ __launch_bounds__(NTHREADS)
void woq_gemm_bf16_wmma(const __hip_bfloat16* __restrict__ Xp,
                        const signed char*     __restrict__ Wp,
                        const signed char*     __restrict__ ZPp,
                        const __hip_bfloat16* __restrict__ Sp,
                        const __hip_bfloat16* __restrict__ Bp,
                        __hip_bfloat16*       __restrict__ OUTp)
{
    const __bf16* X  = reinterpret_cast<const __bf16*>(Xp);
    const int8_t* W  = reinterpret_cast<const int8_t*>(Wp);
    const int8_t* ZP = reinterpret_cast<const int8_t*>(ZPp);
    const __bf16* S  = reinterpret_cast<const __bf16*>(Sp);
    const __bf16* Bb = reinterpret_cast<const __bf16*>(Bp);
    __bf16*       O  = reinterpret_cast<__bf16*>(OUTp);

    // double-buffered stages: 2*(32KB A + 16KB B) = 96KB
    __shared__ __align__(32) __bf16 ldsA[2][TILE_M * KSTAGE];
    __shared__ __align__(32) __bf16 ldsB[2][TILE_N * KSTAGE];

    const int t    = threadIdx.x;
    const int wave = t >> 5;
    const int lane = t & 31;
    const int wm   = wave & 3;          // wave row (M): 0..3
    const int wn   = wave >> 2;         // wave col (N): 0..3
    const int mWave = wm * 64;
    const int nWave = wn * 32;
    const int sel  = lane >> 4;         // lane half selector
    const int l16  = lane & 15;

    const int nBase = blockIdx.x * TILE_N;
    const int mBase = blockIdx.y * TILE_M;

    v8f acc[4][2];
    const v8f zacc = {0.f, 0.f, 0.f, 0.f, 0.f, 0.f, 0.f, 0.f};
#pragma unroll
    for (int mi = 0; mi < 4; ++mi)
#pragma unroll
        for (int ni = 0; ni < 2; ++ni)
            acc[mi][ni] = zacc;

    // ---- async A staging: 256 rows x 64 cols bf16 = 2048 x 16B chunks; 4/thread ----
    const unsigned ldsA0 = (unsigned)(uintptr_t)&ldsA[0][0];
    const unsigned ldsA1 = (unsigned)(uintptr_t)&ldsA[1][0];
    unsigned      aOff[4];
    const __bf16* aPtr[4];
#pragma unroll
    for (int it = 0; it < 4; ++it) {
        const int c   = t + it * NTHREADS;
        const int row = c >> 3;          // 8 chunks (128B) per row
        const int cc  = c & 7;
        aOff[it] = (unsigned)((row * KSTAGE + cc * 8) * 2);
        aPtr[it] = X + (size_t)(mBase + row) * IN_F + cc * 8;
    }
    auto issueA = [&](unsigned base) {
#pragma unroll
        for (int it = 0; it < 4; ++it) {
            const void* gaddr = (const void*)aPtr[it];
            asm volatile("global_load_async_to_lds_b128 %0, %1, off"
                         :: "v"(base + aOff[it]), "v"(gaddr) : "memory");
        }
#pragma unroll
        for (int it = 0; it < 4; ++it) aPtr[it] += KSTAGE;
    };

    // ---- W staging: 128 rows x 64B = 512 x 16B chunks; 1/thread ----
    const int wrow = t >> 2;             // 4 chunks per row
    const int wcc  = t & 3;
    const int8_t* wPtr = W + (size_t)(nBase + wrow) * IN_F + wcc * 16;
    const __bf16* sPtr = S  + (size_t)(nBase + wrow) * NGRP;
    const int8_t* zPtr = ZP + (size_t)(nBase + wrow) * NGRP;
    const int     wLdsIdx = wrow * KSTAGE + wcc * 16;

    auto dequantStore = [&](const int4& raw, float sf, float zs, __bf16* bbuf) {
        const int* rw = (const int*)&raw;
        v16bf dq;
#pragma unroll
        for (int i = 0; i < 16; ++i) {
            const int q = (rw[i >> 2] << ((3 - (i & 3)) * 8)) >> 24;  // sext int8
            dq[i] = (__bf16)__builtin_fmaf((float)q, sf, -zs);
        }
        *(v16bf*)(&bbuf[wLdsIdx]) = dq;
    };

    // ---- prologue: stage 0 into buffer 0 ----
    issueA(ldsA0);
    {
        const int4  raw = *(const int4*)wPtr;  wPtr += KSTAGE;
        const float sf  = (float)sPtr[0];
        const float zs  = (float)zPtr[0] * sf;
        dequantStore(raw, sf, zs, &ldsB[0][0]);
    }
    asm volatile("s_wait_asynccnt 0x0" ::: "memory");
    __syncthreads();

    // ---- pipelined main loop over 64 K-stages ----
    for (int s = 0; s < NSTAGE; ++s) {
        const int buf = s & 1;
        int4  rawN;
        float sfN = 0.f, zsN = 0.f;
        const bool haveNext = (s + 1) < NSTAGE;
        if (haveNext) {
            // issue next stage's DMA + raw weight fetch; latency hidden by compute
            issueA(buf ? ldsA0 : ldsA1);
            rawN = *(const int4*)wPtr;  wPtr += KSTAGE;
            const int gN = (s + 1) >> 1;
            sfN = (float)sPtr[gN];
            zsN = (float)zPtr[gN] * sfN;
            __builtin_prefetch(wPtr, 0, 0);          // global_prefetch_b8, stage s+2
        }

        // ---- compute on current buffer: 2 K-steps of 32, 8 WMMA each ----
        const __bf16* aBuf = &ldsA[buf][0];
        const __bf16* bBuf = &ldsB[buf][0];
#pragma unroll
        for (int kk = 0; kk < 2; ++kk) {
            v16bf afrag[4];
#pragma unroll
            for (int mi = 0; mi < 4; ++mi) {
                const __bf16* base =
                    &aBuf[(mWave + mi * 16 + l16) * KSTAGE + kk * 32 + sel * 16];
                const v8bf lo = *(const v8bf*)(base);
                const v8bf hi = *(const v8bf*)(base + 8);
                afrag[mi] = __builtin_shufflevector(lo, hi,
                    0, 1, 2, 3, 4, 5, 6, 7, 8, 9, 10, 11, 12, 13, 14, 15);
            }
            v16bf bfrag[2];
#pragma unroll
            for (int ni = 0; ni < 2; ++ni) {
                const int nl = nWave + ni * 16 + l16;
                bfrag[ni] = *(const v16bf*)(&bBuf[nl * KSTAGE + kk * 32 + sel * 16]);
            }
#pragma unroll
            for (int mi = 0; mi < 4; ++mi)
#pragma unroll
                for (int ni = 0; ni < 2; ++ni)
                    acc[mi][ni] = __builtin_amdgcn_wmma_f32_16x16x32_bf16(
                        false, afrag[mi], false, bfrag[ni],
                        (short)0, acc[mi][ni], false, false);
        }

        if (haveNext) {
            // finish next stage's weight dequant into the other buffer
            dequantStore(rawN, sfN, zsN, &ldsB[buf ^ 1][0]);
            asm volatile("s_wait_asynccnt 0x0" ::: "memory");
        }
        __syncthreads();
    }

    // ---- epilogue: C layout VGPR r -> rows r (lanes 0-15) / r+8 (lanes 16-31) ----
#pragma unroll
    for (int ni = 0; ni < 2; ++ni) {
        const int col = nBase + nWave + ni * 16 + l16;
        const float bv = (float)Bb[col];
#pragma unroll
        for (int mi = 0; mi < 4; ++mi) {
#pragma unroll
            for (int r = 0; r < 8; ++r) {
                const int row = mBase + mWave + mi * 16 + r + sel * 8;
                O[(size_t)row * OUT_F + col] = (__bf16)(acc[mi][ni][r] + bv);
            }
        }
    }
}

extern "C" void kernel_launch(void* const* d_in, const int* in_sizes, int n_in,
                              void* d_out, int out_size, void* d_ws, size_t ws_size,
                              hipStream_t stream) {
    (void)in_sizes; (void)n_in; (void)out_size; (void)d_ws; (void)ws_size;
    const __hip_bfloat16* x  = (const __hip_bfloat16*)d_in[0];
    const signed char*    w  = (const signed char*)d_in[1];
    const signed char*    zp = (const signed char*)d_in[2];
    const __hip_bfloat16* s  = (const __hip_bfloat16*)d_in[3];
    const __hip_bfloat16* b  = (const __hip_bfloat16*)d_in[4];
    __hip_bfloat16* out = (__hip_bfloat16*)d_out;

    dim3 grid(OUT_F / TILE_N, MTOK / TILE_M);  // 86 x 16
    dim3 block(NTHREADS);
    woq_gemm_bf16_wmma<<<grid, block, 0, stream>>>(x, w, zp, s, b, out);
}
